// KnnModule_56873956933714
// MI455X (gfx1250) — compile-verified, matrix operand order checked by reading, not run on hardware
//
#include <hip/hip_runtime.h>
#include <hip/hip_bf16.h>
#include <math.h>

typedef __attribute__((ext_vector_type(16))) _Float16     v16h;
typedef __attribute__((ext_vector_type(8)))  float        v8f;
typedef __attribute__((ext_vector_type(4)))  unsigned int u32x4;
typedef __attribute__((ext_vector_type(8)))  unsigned int u32x8;

#define B_ROWS      2048
#define DIM         1024
#define NUM_CLASSES 1000
#define KMAX        200
#define TM          128
#define TN          128
#define TK          32
#define LDH         40                   // LDS row pitch in halfs (80B, conflict-free b128)
#define BUFBYTES    (TM * LDH * 2)       // 10240 bytes per tile buffer

__device__ __forceinline__ unsigned fkey(float f) {
    unsigned u = __float_as_uint(f);
    return (u & 0x80000000u) ? ~u : (u | 0x80000000u);
}

// ---------------------------------------------------------------------------
// TDM: issue a 2D tensor_load_to_lds of a TM x TK (rows x halfs) tile.
// Descriptor packed per CDNA5 ISA.
// Hardware pad: +16B after every 64B row -> LDS pitch = 80B = LDH halfs.
// ---------------------------------------------------------------------------
__device__ __forceinline__ void tdm_load_tile(unsigned lds_addr, const _Float16* gptr)
{
    const unsigned long long ga = (unsigned long long)(size_t)gptr;
    u32x4 g0;
    g0[0] = 1u;                                            // count=1, user mode
    g0[1] = lds_addr;                                      // lds_addr [63:32]
    g0[2] = (unsigned)(ga & 0xFFFFFFFFu);                  // global_addr[31:0]
    g0[3] = (unsigned)((ga >> 32) & 0x01FFFFFFu)           // global_addr[56:32]
          | (2u << 30);                                    // type=2 ("image")
    u32x8 g1;
    g1[0] = (1u << 16)        // data_size = 1 -> 2 bytes
          | (1u << 20)        // pad_enable
          | (3u << 22)        // pad_interval: 16 DWORDs (64B) stored per pad
          | (3u << 25);       // pad_amount: 4 DWORDs (16B)
    g1[1] = ((unsigned)TK & 0xFFFFu) << 16;                // tensor_dim0[15:0] = 32
    g1[2] = ((unsigned)TM & 0xFFFFu) << 16;                // dim0[31:16]=0 | tensor_dim1[15:0]=128
    g1[3] = ((unsigned)TK & 0xFFFFu) << 16;                // dim1[31:16]=0 | tile_dim0=32
    g1[4] = (unsigned)TM;                                  // tile_dim1=128 | tile_dim2=0
    g1[5] = (unsigned)DIM;                                 // tensor_dim0_stride[31:0] = 1024 elems
    g1[6] = 0u;                                            // stride[47:32] | dim1_stride[15:0]
    g1[7] = 0u;                                            // dim1_stride[47:16]
    asm volatile("tensor_load_to_lds %0, %1" :: "s"(g0), "s"(g1) : "memory");
}

// ---------------------------------------------------------------------------
// f32 -> f16 converters (feed the TDM with pre-converted, pre-padded f16 data)
// ---------------------------------------------------------------------------
__global__ __launch_bounds__(256) void cvt_feats(
    const float* __restrict__ src, _Float16* __restrict__ dst, int n)
{
    const int i = (blockIdx.x * 256 + threadIdx.x) * 8;
    if (i >= n) return;
    float4 a = *(const float4*)(src + i);
    float4 b = *(const float4*)(src + i + 4);
    union { _Float16 h[8]; uint4 u; } p;
    p.h[0] = (_Float16)a.x; p.h[1] = (_Float16)a.y; p.h[2] = (_Float16)a.z; p.h[3] = (_Float16)a.w;
    p.h[4] = (_Float16)b.x; p.h[5] = (_Float16)b.y; p.h[6] = (_Float16)b.z; p.h[7] = (_Float16)b.w;
    *(uint4*)(dst + i) = p.u;
}

__global__ __launch_bounds__(256) void cvt_chunk(
    const float* __restrict__ train, _Float16* __restrict__ dst,
    int start, int ncValid, int rowsTile)
{
    const int e = (blockIdx.x * 256 + threadIdx.x) * 8;
    const int r = e >> 10;          // row within chunk
    const int c = e & (DIM - 1);
    if (r >= rowsTile) return;
    union { _Float16 h[8]; uint4 u; } p;
    if (r < ncValid) {
        const float* src = train + (size_t)(start + r) * DIM + c;
        float4 a = *(const float4*)(src);
        float4 b = *(const float4*)(src + 4);
        p.h[0] = (_Float16)a.x; p.h[1] = (_Float16)a.y; p.h[2] = (_Float16)a.z; p.h[3] = (_Float16)a.w;
        p.h[4] = (_Float16)b.x; p.h[5] = (_Float16)b.y; p.h[6] = (_Float16)b.z; p.h[7] = (_Float16)b.w;
    } else {
        #pragma unroll
        for (int q = 0; q < 8; ++q) p.h[q] = (_Float16)0.0f;
    }
    *(uint4*)(dst + (size_t)r * DIM + c) = p.u;
}

// ---------------------------------------------------------------------------
// GEMM: sim[b, j] = dot(featsH[b,:], chunkH[j,:])  — TDM-staged, double-buffered
// ---------------------------------------------------------------------------
__global__ __launch_bounds__(256) void knn_gemm(
    const _Float16* __restrict__ Ah,   // (2048, 1024) f16
    const _Float16* __restrict__ Ch,   // (ncPad, 1024) f16, zero-padded rows
    float* __restrict__ sim,           // (2048, ncPad) f32
    int ncPad)
{
    __shared__ __align__(16) _Float16 As[2][TM * LDH];
    __shared__ __align__(16) _Float16 Bs[2][TN * LDH];

    const int tid   = threadIdx.x;
    const int lane  = tid & 31;
    const int w     = tid >> 5;       // wave 0..7
    const int waveM = w >> 1;         // 0..3 -> 32-row band
    const int waveN = w & 1;          // 0..1 -> 64-col band
    const int l15   = lane & 15;
    const int hsel  = lane >> 4;
    const int rowBlock = blockIdx.y * TM;
    const int colBlock = blockIdx.x * TN;

    const unsigned ldsA = (unsigned)(size_t)(void*)&As[0][0];
    const unsigned ldsB = (unsigned)(size_t)(void*)&Bs[0][0];
    const _Float16* aBase = Ah + (size_t)rowBlock * DIM;
    const _Float16* bBase = Ch + (size_t)colBlock * DIM;

    v8f acc[2][4];
    #pragma unroll
    for (int mt = 0; mt < 2; ++mt)
        #pragma unroll
        for (int nt = 0; nt < 4; ++nt)
            #pragma unroll
            for (int r = 0; r < 8; ++r) acc[mt][nt][r] = 0.0f;

    const int nSteps = DIM / TK;      // 32

    // prime buffer 0 (wave 0 drives the TDM; EXEC/other waves uninvolved)
    if (w == 0) {
        tdm_load_tile(ldsA, aBase);
        tdm_load_tile(ldsB, bBase);
    }

    for (int s = 0; s < nSteps; ++s) {
        const int cur = s & 1;
        if (w == 0) __builtin_amdgcn_s_wait_tensorcnt(0);
        __syncthreads();   // releases readers of buf[cur]; fences writers of buf[cur^1]

        if (w == 0 && (s + 1) < nSteps) {
            const int k1 = (s + 1) * TK;
            const unsigned boff = (unsigned)((cur ^ 1) * BUFBYTES);
            tdm_load_tile(ldsA + boff, aBase + k1);
            tdm_load_tile(ldsB + boff, bBase + k1);
        }

        // ---- fragments per ISA 16-bit A 16x32 layout ----
        v16h aF[2], bF[4];
        #pragma unroll
        for (int mt = 0; mt < 2; ++mt) {
            union { uint4 u4[2]; v16h v; } fa;
            const int base = (waveM*32 + mt*16 + l15) * LDH;
            fa.u4[0] = *(const uint4*)&As[cur][base + hsel*8];
            fa.u4[1] = *(const uint4*)&As[cur][base + 16 + hsel*8];
            aF[mt] = fa.v;
        }
        #pragma unroll
        for (int nt = 0; nt < 4; ++nt) {
            union { uint4 u4[2]; v16h v; } fb;
            const int base = (waveN*64 + nt*16 + l15) * LDH;
            fb.u4[0] = *(const uint4*)&Bs[cur][base + hsel*8];
            fb.u4[1] = *(const uint4*)&Bs[cur][base + 16 + hsel*8];
            bF[nt] = fb.v;
        }
        #pragma unroll
        for (int mt = 0; mt < 2; ++mt)
            #pragma unroll
            for (int nt = 0; nt < 4; ++nt)
                acc[mt][nt] = __builtin_amdgcn_wmma_f32_16x16x32_f16(
                    false, aF[mt], false, bF[nt], (short)0, acc[mt][nt], false, false);
    }

    #pragma unroll
    for (int mt = 0; mt < 2; ++mt)
        #pragma unroll
        for (int nt = 0; nt < 4; ++nt)
            #pragma unroll
            for (int r = 0; r < 8; ++r) {
                const int row = rowBlock + waveM*32 + mt*16 + hsel*8 + r;
                const int col = colBlock + waveN*64 + nt*16 + l15;
                sim[(size_t)row * ncPad + col] = acc[mt][nt][r];
            }
}

// ---------------------------------------------------------------------------
// Init running top lists
// ---------------------------------------------------------------------------
__global__ void knn_init(float* __restrict__ topV, int* __restrict__ topI) {
    int i = blockIdx.x * blockDim.x + threadIdx.x;
    if (i < B_ROWS * KMAX) { topV[i] = -__builtin_inff(); topI[i] = 0; }
}

// ---------------------------------------------------------------------------
// Merge: exact top-200 of (chunk sims U old top list) via 4x8-bit radix select
// ---------------------------------------------------------------------------
__global__ __launch_bounds__(256) void knn_merge(
    const float* __restrict__ sim, int chunkStart, int ncValid, int ncPad,
    float* __restrict__ topV, int* __restrict__ topI)
{
    const int row = blockIdx.x;
    const int tid = threadIdx.x;
    const float* simRow = sim + (size_t)row * ncPad;

    __shared__ float    oldV[KMAX];
    __shared__ int      oldI[KMAX];
    __shared__ unsigned hist[256];
    __shared__ int      sSel, sRem;
    __shared__ unsigned sCnt;

    for (int i = tid; i < KMAX; i += 256) {
        oldV[i] = topV[row*KMAX + i];
        oldI[i] = topI[row*KMAX + i];
    }
    if (tid == 0) sRem = KMAX;
    __syncthreads();

    const int M = ncValid + KMAX;
    unsigned prefix = 0, pmask = 0;

    for (int pass = 3; pass >= 0; --pass) {
        const int shift = pass * 8;
        hist[tid] = 0;
        __syncthreads();
        for (int i = tid; i < M; i += 256) {
            const float v = (i < ncValid) ? simRow[i] : oldV[i - ncValid];
            const unsigned u = fkey(v);
            if ((u & pmask) == prefix) atomicAdd(&hist[(u >> shift) & 255], 1u);
        }
        __syncthreads();
        if (tid == 0) {
            int b = 255, r = sRem;
            while (b > 0) { int c = (int)hist[b]; if (c >= r) break; r -= c; --b; }
            sSel = b; sRem = r;
        }
        __syncthreads();
        prefix |= ((unsigned)sSel) << shift;
        pmask  |= 0xFFu << shift;
        __syncthreads();
    }

    if (tid == 0) sCnt = 0;
    __syncthreads();
    const unsigned t = prefix;

    for (int i = tid; i < M; i += 256) {
        float v; int g;
        if (i < ncValid) { v = simRow[i]; g = chunkStart + i; }
        else             { v = oldV[i - ncValid]; g = oldI[i - ncValid]; }
        if (fkey(v) > t) {
            const unsigned p = atomicAdd(&sCnt, 1u);
            topV[row*KMAX + p] = v;
            topI[row*KMAX + p] = g;
        }
    }
    __syncthreads();

    if (tid == 0) {
        int p = (int)sCnt;
        for (int i = 0; i < KMAX && p < KMAX; ++i) {
            if (fkey(oldV[i]) == t) { topV[row*KMAX+p] = oldV[i]; topI[row*KMAX+p] = oldI[i]; ++p; }
        }
        for (int i = 0; i < ncValid && p < KMAX; ++i) {
            const float v = simRow[i];
            if (fkey(v) == t) { topV[row*KMAX+p] = v; topI[row*KMAX+p] = chunkStart + i; ++p; }
        }
    }
}

// ---------------------------------------------------------------------------
// Finalize: sort top-200 desc, softmax(/T) over 200, prefix-k class votes
// ---------------------------------------------------------------------------
__global__ __launch_bounds__(256) void knn_final(
    const float* __restrict__ topV, const int* __restrict__ topI,
    const int* __restrict__ labels, float* __restrict__ out)
{
    const int row = blockIdx.x;
    const int tid = threadIdx.x;

    __shared__ float v[256];
    __shared__ int   id[256];
    __shared__ float wgt[256];
    __shared__ float cls[NUM_CLASSES];
    __shared__ float sSum;

    if (tid < KMAX) { v[tid] = topV[row*KMAX + tid]; id[tid] = topI[row*KMAX + tid]; }
    else            { v[tid] = -__builtin_inff();    id[tid] = 0x7FFFFFFF; }
    __syncthreads();

    for (int k = 2; k <= 256; k <<= 1) {
        for (int j = k >> 1; j > 0; j >>= 1) {
            const int ixj = tid ^ j;
            if (ixj > tid) {
                const bool dirDesc = ((tid & k) == 0);
                const float va = v[tid], vb = v[ixj];
                const int   ia = id[tid], ib = id[ixj];
                const bool aBeforeB = (va > vb) || (va == vb && ia < ib);
                if (dirDesc ? !aBeforeB : aBeforeB) {
                    v[tid] = vb; v[ixj] = va;
                    id[tid] = ib; id[ixj] = ia;
                }
            }
            __syncthreads();
        }
    }

    const float vmax = v[0];
    wgt[tid] = (tid < KMAX) ? expf((v[tid] - vmax) * (1.0f / 0.07f)) : 0.0f;
    __syncthreads();
    if (tid == 0) {
        float s = 0.0f;
        for (int i = 0; i < KMAX; ++i) s += wgt[i];
        sSum = s;
    }
    __syncthreads();
    if (tid < KMAX) wgt[tid] /= sSum;

    for (int c = tid; c < NUM_CLASSES; c += 256) cls[c] = 0.0f;
    __syncthreads();

    const int ks[4] = {10, 20, 100, 200};
    int prev = 0;
    for (int kk = 0; kk < 4; ++kk) {
        if (tid == 0) {
            for (int i = prev; i < ks[kk]; ++i) cls[labels[id[i]]] += wgt[i];
        }
        __syncthreads();
        float* dst = out + ((size_t)kk * B_ROWS + row) * NUM_CLASSES;
        for (int c = tid; c < NUM_CLASSES; c += 256) dst[c] = cls[c];
        __syncthreads();
        prev = ks[kk];
    }
}

// ---------------------------------------------------------------------------
extern "C" void kernel_launch(void* const* d_in, const int* in_sizes, int n_in,
                              void* d_out, int out_size, void* d_ws, size_t ws_size,
                              hipStream_t stream)
{
    (void)n_in; (void)out_size;
    const float* feats  = (const float*)d_in[0];   // (2048, 1024)
    const float* train  = (const float*)d_in[1];   // (N, 1024)
    const int*   labels = (const int*)d_in[2];     // (N,)
    float*       out    = (float*)d_out;           // 4 x (2048, 1000)
    const int N = in_sizes[1] / DIM;

    // workspace layout: topV | topI | featsH | chunkH | sim
    char* ws = (char*)d_ws;
    float*    topV   = (float*)ws;
    int*      topI   = (int*)(ws + (size_t)B_ROWS * KMAX * 4);
    _Float16* featsH = (_Float16*)(ws + (size_t)B_ROWS * KMAX * 8);
    const size_t fixedBytes = (size_t)B_ROWS * KMAX * 8 + (size_t)B_ROWS * DIM * 2;

    // per padded column: 1024 f16 (chunkH) + 2048 f32 (sim column) = 10240 B
    const size_t varBytes = (ws_size > fixedBytes) ? (ws_size - fixedBytes) : 0;
    int ncPad = (int)(varBytes / 10240);
    ncPad = (ncPad / TN) * TN;
    if (ncPad > 8192) ncPad = 8192;
    if (ncPad < TN)   ncPad = TN;

    _Float16* chunkH = featsH + (size_t)B_ROWS * DIM;
    float*    sim    = (float*)((char*)chunkH + (size_t)ncPad * DIM * 2);

    knn_init<<<(B_ROWS * KMAX + 255) / 256, 256, 0, stream>>>(topV, topI);
    cvt_feats<<<(B_ROWS * DIM / 8 + 255) / 256, 256, 0, stream>>>(feats, featsH, B_ROWS * DIM);

    for (int start = 0; start < N; start += ncPad) {
        int ncValid = N - start;
        if (ncValid > ncPad) ncValid = ncPad;
        const int tilesX = (ncValid + TN - 1) / TN;
        const int rowsTile = tilesX * TN;

        cvt_chunk<<<(rowsTile * (DIM / 8) + 255) / 256, 256, 0, stream>>>(
            train, chunkH, start, ncValid, rowsTile);

        dim3 grid(tilesX, B_ROWS / TM);
        knn_gemm<<<grid, 256, 0, stream>>>(featsH, chunkH, sim, ncPad);

        knn_merge<<<B_ROWS, 256, 0, stream>>>(sim, start, ncValid, ncPad, topV, topI);
    }

    knn_final<<<B_ROWS, 256, 0, stream>>>(topV, topI, labels, out);
}
